// MACEScoreNetwork_24550033064016
// MI455X (gfx1250) — compile-verified, matrix operand order checked by reading, not run on hardware
//
#include <hip/hip_runtime.h>
#include <hip/hip_bf16.h>
#include <math.h>

// ---------------------------------------------------------------------------
// MACE score network on MI455X (gfx1250, wave32).
//  - All GEMMs via v_wmma_f32_16x16x32_f16 (f16 in, f32 accumulate).
//  - Edge radial MLPs fused with the j-reduction per (b,i); per-edge "we"
//    tensors (335MB if materialized) never leave LDS (320KB/WGP).
//  - Weights converted f32->f16 and pre-swizzled into WMMA B-fragment order
//    once; per-workgroup staging is a TDM tensor_load_to_lds (s_wait_tensorcnt)
//    or plain b128 copies. Fragment loads are ds_load_b128 pairs.
// ---------------------------------------------------------------------------

typedef _Float16 h16;
typedef __attribute__((ext_vector_type(16))) _Float16 v16h;
typedef __attribute__((ext_vector_type(8)))  _Float16 v8h;
typedef __attribute__((ext_vector_type(8)))  float    v8f;
typedef __attribute__((ext_vector_type(4)))  unsigned int u32x4;
typedef __attribute__((ext_vector_type(8)))  int      i32x8;
typedef __attribute__((ext_vector_type(4)))  int      i32x4;

#define DEV __device__ __forceinline__

static constexpr int BB = 32, NN = 64, KD = 128, NBES = 8, HML = 512;
static constexpr int M_EDGE = BB * NN * NN;   // 131072 edges
static constexpr int NNODE  = BB * NN;        // 2048 nodes
static constexpr int FPAD   = 672;            // 641 feature cols padded to 21*32
static constexpr float RMAX = 5.0f;
static constexpr float INV_AVG = 1.0f / 16.0f;

// swizzled f16 weight blob sizes (halves)
static constexpr int R1W0 = 32 * 64, RW1 = 64 * 64, RW2 = 64 * 64;
static constexpr int R1W3 = 64 * 384, R2W3 = 64 * 256;
static constexpr int R1_BLOB = R1W0 + RW1 + RW2 + R1W3;   // 34816
static constexpr int R2_BLOB = R1W0 + RW1 + RW2 + R2W3;   // 26624

DEV int lane_id() { return threadIdx.x & 31; }
DEV int wave_id() { return threadIdx.x >> 5; }
DEV float fast_rcp(float x) { return __builtin_amdgcn_rcpf(x); }
DEV float silu_f(float x) { return x * fast_rcp(1.0f + __expf(-x)); }
DEV v8f v8f_zero() { v8f z = {0.f,0.f,0.f,0.f,0.f,0.f,0.f,0.f}; return z; }

// ---- WMMA 16x16x32 f16 fragment helpers (wave32 lane layouts per ISA) ----
DEV int frag_k_of(int e, int klo) {
  return (e < 8) ? (klo + e) : (klo + 16 + (e - 8));
}
DEV v16h cat16(v8h lo, v8h hi) {
  v16h r;
#pragma unroll
  for (int e = 0; e < 8; ++e) { r[e] = lo[e]; r[8 + e] = hi[e]; }
  return r;
}
// A: 16 x (32*chunks) tile, row-major in LDS, 16B-aligned rows -> 2x ds_load_b128
DEV v16h frag_a_vec(const h16* A, int lda) {
  int l = lane_id(), m = l & 15, klo = (l < 16) ? 0 : 8;
  const h16* p = A + m * lda + klo;
  return cat16(*(const v8h*)p, *(const v8h*)(p + 16));
}
// B: pre-swizzled tile (512 halves: [32 lanes][16 elems]) -> 2x ds_load_b128
DEV v16h frag_b_swz(const h16* tile) {
  const h16* p = tile + lane_id() * 16;
  return cat16(*(const v8h*)p, *(const v8h*)(p + 8));
}
DEV v8f wmma_f16(v16h a, v16h b, v8f c) {
  return __builtin_amdgcn_wmma_f32_16x16x32_f16(false, a, false, b, (short)0, c,
                                                false, false);
}
// GEMM of one 16x16 tile against swizzled weights: Wsw tiles of 512 halves,
// tile index (kc*nct + ct).
DEV v8f gemm_swz(v8f acc, const h16* A, int lda, const h16* Wsw, int nct, int ct,
                 int Ksz) {
  for (int kc = 0; kc < (Ksz >> 5); ++kc)
    acc = wmma_f16(frag_a_vec(A + kc * 32, lda),
                   frag_b_swz(Wsw + ((kc * nct) + ct) * 512), acc);
  return acc;
}

// ---------------------------------------------------------------------------
// TDM: stage a contiguous byte blob global -> LDS (1-row 2D tile, 8B elems).
// 5-arg builtin on ROCm7.2/clang-22, 6-arg on amdgpu-toolchain/clang-23.
// ---------------------------------------------------------------------------
DEV void tdm_copy_to_lds(const void* gsrc, void* ldst, unsigned bytes) {
  unsigned long long ga = (unsigned long long)(size_t)gsrc;
  unsigned lds_addr = (unsigned)(size_t)ldst;
  unsigned n8 = bytes >> 3;  // 8-byte units; fits tile_dim0 (16b) for our blobs
  u32x4 g0;
  g0[0] = 1u;                                   // count=1 valid descriptor
  g0[1] = lds_addr;                             // lds_addr
  g0[2] = (unsigned)ga;                         // global_addr[31:0]
  g0[3] = (unsigned)((ga >> 32) & 0x1FFFFFFu) | (2u << 30);  // addr hi | type=2
  i32x8 g1;
  g1[0] = (int)(3u << 16);                      // workgroup_mask=0, data_size=8B
  g1[1] = (int)((n8 & 0xFFFFu) << 16);          // tensor_dim0 [63:48]
  g1[2] = (int)(((n8 >> 16) & 0xFFFFu) | (1u << 16)); // dim0 hi | tensor_dim1=1
  g1[3] = (int)((n8 & 0xFFFFu) << 16);          // tile_dim0 [127:112]
  g1[4] = 1;                                    // tile_dim1=1, tile_dim2=0
  g1[5] = (int)n8;                              // tensor_dim0_stride lo32
  g1[6] = (int)((n8 & 0xFFFFu) << 16);          // stride0 hi=0 | stride1 lo16
  g1[7] = (int)(n8 >> 16);                      // stride1 hi
  i32x4 z4 = {0, 0, 0, 0};
#if __clang_major__ >= 23
  i32x8 z8 = {0, 0, 0, 0, 0, 0, 0, 0};
  __builtin_amdgcn_tensor_load_to_lds(g0, g1, z4, z4, z8, 0);
#else
  __builtin_amdgcn_tensor_load_to_lds(g0, g1, z4, z4, 0);
#endif
}

// 64-row FC layer on LDS activations; blockDim = 256 (8 waves), uniform tiles.
// blayout=0: row-major output (A-consumable). blayout=1: B-fragment-swizzled.
DEV void run_fc_layer(const h16* actIn, int ldin, int Ksz,
                      const h16* Wsw, int Ncols, const float* bias,
                      h16* actOut, int ldout, int mode /*0:none 1:silu*/,
                      const float* rowScale, int blayout) {
  int nct = Ncols >> 4, ntiles = 4 * nct;
  int l = lane_id(), n = l & 15, mb = (l < 16) ? 0 : 8;
  for (int t = wave_id(); t < ntiles; t += 8) {
    int rt = t / nct, ct = t % nct;
    v8f acc = v8f_zero();
    acc = gemm_swz(acc, actIn + rt * 16 * ldin, ldin, Wsw, nct, ct, Ksz);
#pragma unroll
    for (int v = 0; v < 8; ++v) {
      int row = rt * 16 + mb + v, col = ct * 16 + n;
      float x = acc[v] + bias[col];
      if (mode == 1) x = silu_f(x);
      if (rowScale) x *= rowScale[row];
      if (blayout == 0) {
        actOut[row * ldout + col] = (h16)x;
      } else {
        // scatter into B-fragment order: row is the K index of the next GEMM
        int kc = row >> 5, rr = row & 31;
        int e2 = (rr & 16) ? (8 + (rr & 7)) : (rr & 7);
        int lb = (col & 15) + ((rr & 8) ? 16 : 0);
        actOut[((kc * nct) + (col >> 4)) * 512 + lb * 16 + e2] = (h16)x;
      }
    }
  }
  __syncthreads();
}

// ---------------------------------------------------------------------------
// Kernel 0: convert one f32 weight matrix to f16 in B-fragment-swizzled order.
// ---------------------------------------------------------------------------
__global__ void k_swz(h16* __restrict__ dst, const float* __restrict__ src,
                      int Ksz, int Krows, int Ncols) {
  int idx = blockIdx.x * 256 + threadIdx.x;
  if (idx >= Ksz * Ncols) return;
  int nct = Ncols >> 4;
  int e = idx & 15, l = (idx >> 4) & 31, tile = idx >> 9;
  int ct = tile % nct, kc = tile / nct;
  int klo = (l & 16) ? 8 : 0, n = l & 15;
  int k = kc * 32 + frag_k_of(e, klo);
  dst[idx] = (k < Krows) ? (h16)src[k * Ncols + ct * 16 + n] : (h16)0.0f;
}

// ---------------------------------------------------------------------------
// Kernel 1: periodic geometry, Bessel*cutoff basis (f16 cache), Ylm, mask.
// ---------------------------------------------------------------------------
__global__ void k_geom(const float* __restrict__ pos, const float* __restrict__ cell,
                       h16* __restrict__ radf, float* __restrict__ yv,
                       float* __restrict__ maskv) {
  int e = blockIdx.x * 256 + threadIdx.x;
  if (e >= M_EDGE) return;
  int b = e >> 12, rem = e & 4095, i = rem >> 6, j = rem & 63;
  float d[3];
#pragma unroll
  for (int c = 0; c < 3; ++c) {
    float dd = pos[(b * NN + i) * 3 + c] - pos[(b * NN + j) * 3 + c];
    d[c] = dd - rintf(dd);
  }
  float dc[3];
#pragma unroll
  for (int c = 0; c < 3; ++c)
    dc[c] = d[0] * cell[b * 9 + 0 + c] + d[1] * cell[b * 9 + 3 + c] +
            d[2] * cell[b * 9 + 6 + c];
  float r2 = fmaxf(dc[0] * dc[0] + dc[1] * dc[1] + dc[2] * dc[2], 1e-12f);
  float r = sqrtf(r2);
  bool mk = (r < RMAX) && (i != j);
  float rs = mk ? r : 1.0f;
  float inv_r = fast_rcp(rs);
  float mf = mk ? 1.0f : 0.0f;
  float ux = dc[0] * inv_r * mf, uy = dc[1] * inv_r * mf, uz = dc[2] * inv_r * mf;
  float u = rs * (1.0f / RMAX), u2 = u * u, u5 = u2 * u2 * u;
  float fc = 1.0f - 21.0f * u5 + 35.0f * u5 * u - 15.0f * u5 * u2;
  fc = (u < 1.0f) ? fc : 0.0f;
  float fcm = fc * mf;
  float pref = sqrtf(2.0f / RMAX) * inv_r;
#pragma unroll
  for (int nb = 1; nb <= NBES; ++nb) {
    float rb = pref * __sinf((float)nb * ((float)M_PI / RMAX) * rs);
    radf[e * NBES + (nb - 1)] = (h16)(rb * fcm);
  }
  const float s3 = 1.7320508075688772f, s5 = 2.2360679774997896f,
              s15 = 3.8729833462074170f;
  yv[e * 9 + 0] = 1.0f;
  yv[e * 9 + 1] = s3 * ux; yv[e * 9 + 2] = s3 * uy; yv[e * 9 + 3] = s3 * uz;
  yv[e * 9 + 4] = s15 * ux * uy;
  yv[e * 9 + 5] = s15 * uy * uz;
  yv[e * 9 + 6] = 0.5f * s5 * (3.0f * uz * uz - 1.0f);
  yv[e * 9 + 7] = s15 * ux * uz;
  yv[e * 9 + 8] = 0.5f * s15 * (ux * ux - uy * uy);
  maskv[e] = mf;
}

// ---------------------------------------------------------------------------
// Kernel 2: per-(b,i) fused radial MLP #1 (WMMA chain in LDS, weights staged
// via TDM) + WMMA j-reduction against Ylm -> msg[b,i,m(9),k].
// ---------------------------------------------------------------------------
__global__ void k_edge_pass1(
    const h16* __restrict__ radf, const float* __restrict__ yv,
    const float* __restrict__ maskv, const h16* __restrict__ wblob,
    const float* __restrict__ b0g, const float* __restrict__ b1g,
    const float* __restrict__ b2g, const float* __restrict__ b3g,
    const float* __restrict__ wembed, float* __restrict__ msg) {
  extern __shared__ char smem[];
  float* b0 = (float*)smem; float* b1 = b0 + 64; float* b2 = b1 + 64;
  float* b3 = b2 + 64; float* mk = b3 + 384;                  // 640 floats
  h16* w0 = (h16*)(mk + 64);                                   // blob dest
  h16* w1 = w0 + R1W0; h16* w2 = w1 + RW1; h16* w3 = w2 + RW2;
  h16* a0 = w0 + R1_BLOB;
  h16* hA = a0 + 64 * 32; h16* hB = hA + 64 * 64;
  h16* we = hB + 64 * 64;                                      // B-swizzled
  h16* yA = we + 64 * 384;                                     // 3 x [16x64]

  int tid = threadIdx.x;
  int row = blockIdx.x;        // b*N + i
  int eb = row * NN;

  if (tid == 0) tdm_copy_to_lds(wblob, w0, (unsigned)(R1_BLOB * 2));

  for (int idx = tid; idx < 64; idx += 256) {
    b0[idx] = b0g[idx]; b1[idx] = b1g[idx]; b2[idx] = b2g[idx];
    mk[idx] = maskv[eb + idx];
  }
  for (int idx = tid; idx < 384; idx += 256) b3[idx] = b3g[idx];
  for (int idx = tid; idx < 64 * 4; idx += 256) {   // a0: [64][32], zero-padded
    int j = idx >> 2, q = idx & 3;
    v8h z = {};
    ((v8h*)a0)[idx] = (q == 0) ? *(const v8h*)(radf + (eb + j) * NBES) : z;
  }
  for (int idx = tid; idx < 3 * 16 * 64; idx += 256) {
    int g = idx >> 10, rem = idx & 1023, m = rem >> 6, j = rem & 63;
    int mg = (g == 0) ? 1 : (g == 1) ? 3 : 5;
    int go = (g == 2) ? 4 : g;
    yA[idx] = (m < mg) ? (h16)yv[(eb + j) * 9 + go + m] : (h16)0.0f;
  }
  __builtin_amdgcn_s_wait_tensorcnt(0);
  __syncthreads();

  run_fc_layer(a0, 32, 32, w0, 64, b0, hA, 64, 1, nullptr, 0);
  run_fc_layer(hA, 64, 64, w1, 64, b1, hB, 64, 1, nullptr, 0);
  run_fc_layer(hB, 64, 64, w2, 64, b2, hA, 64, 1, nullptr, 0);
  run_fc_layer(hA, 64, 64, w3, 384, b3, we, 0, 0, mk, 1);  // we1 masked, B-swz

  // msg[m,k] = (w_embed[k]/16) * sum_j Y_g[m,j] * we[j, g*128+k]  (K-dim = j)
  int l = lane_id(), n = l & 15, mb = (l < 16) ? 0 : 8;
  for (int t = wave_id(); t < 24; t += 8) {
    int g = t >> 3, ct = t & 7;
    int mg = (g == 0) ? 1 : (g == 1) ? 3 : 5;
    int go = (g == 2) ? 4 : g;
    v8f acc = v8f_zero();
    acc = gemm_swz(acc, yA + g * 16 * 64, 64, we, 24, g * 8 + ct, 64);
#pragma unroll
    for (int v = 0; v < 8; ++v) {
      int m = mb + v;
      if (m < mg) {
        int k = ct * 16 + n;
        msg[(row * 9 + go + m) * KD + k] = acc[v] * wembed[k] * INV_AVG;
      }
    }
  }
}

// ---------------------------------------------------------------------------
// Kernel 3: node mixes  s1 = msg0@l0 + (sum_m msg2^2)@l2 ; v1 = msg1@l1.
// Mix weights staged via TDM (pre-swizzled blob).
// ---------------------------------------------------------------------------
__global__ void k_mix1(const float* __restrict__ msg, const h16* __restrict__ mixblob,
                       float* __restrict__ s1g, float* __restrict__ h1g,
                       float* __restrict__ v1g) {
  extern __shared__ char smem[];
  h16* l0 = (h16*)smem; h16* l1 = l0 + KD * KD; h16* l2 = l1 + KD * KD;
  h16* A0 = l2 + KD * KD; h16* Aq = A0 + 64 * KD; h16* A1 = Aq + 64 * KD;
  int tid = threadIdx.x, rb = blockIdx.x * 64;
  if (tid == 0) tdm_copy_to_lds(mixblob, l0, (unsigned)(3 * KD * KD * 2));
  for (int idx = tid; idx < 64 * KD; idx += 256) {
    int r = idx >> 7, c = idx & 127;
    A0[idx] = (h16)msg[((rb + r) * 9 + 0) * KD + c];
    float q = 0.f;
#pragma unroll
    for (int m = 4; m < 9; ++m) { float z = msg[((rb + r) * 9 + m) * KD + c]; q += z * z; }
    Aq[idx] = (h16)q;
  }
  for (int idx = tid; idx < 192 * KD; idx += 256) {
    int rr = idx >> 7, c = idx & 127, r = rr / 3, m = rr % 3;
    A1[idx] = (h16)msg[((rb + r) * 9 + 1 + m) * KD + c];
  }
  __builtin_amdgcn_s_wait_tensorcnt(0);
  __syncthreads();
  int l = lane_id(), n = l & 15, mb = (l < 16) ? 0 : 8;
  for (int t = wave_id(); t < 128; t += 8) {
    if (t < 32) {  // s1 tiles: 4x8
      int rt = t >> 3, ct = t & 7;
      v8f acc = v8f_zero();
      acc = gemm_swz(acc, A0 + rt * 16 * KD, KD, l0, 8, ct, KD);
      acc = gemm_swz(acc, Aq + rt * 16 * KD, KD, l2, 8, ct, KD);
#pragma unroll
      for (int v = 0; v < 8; ++v) {
        int row = rb + rt * 16 + mb + v, col = ct * 16 + n;
        float x = acc[v];
        s1g[row * KD + col] = x;
        h1g[row * KD + col] = silu_f(x);
      }
    } else {       // v1 tiles: 12x8 (192 rows = 3 per node)
      int t2 = t - 32, rt = t2 >> 3, ct = t2 & 7;
      v8f acc = v8f_zero();
      acc = gemm_swz(acc, A1 + rt * 16 * KD, KD, l1, 8, ct, KD);
#pragma unroll
      for (int v = 0; v < 8; ++v) {
        int rr = rt * 16 + mb + v, col = ct * 16 + n, r = rr / 3, m = rr % 3;
        v1g[((rb + r) * 3 + m) * KD + col] = acc[v];
      }
    }
  }
}

// ---------------------------------------------------------------------------
// Kernel 4: per-(b,i) radial MLP #2 (WMMA, TDM-staged weights) + VALU
// reduction over j for msg_a/msg_b -> s2pre.
// ---------------------------------------------------------------------------
__global__ void k_pass2(
    const h16* __restrict__ radf, const float* __restrict__ yv,
    const float* __restrict__ maskv, const h16* __restrict__ wblob,
    const float* __restrict__ b0g, const float* __restrict__ b1g,
    const float* __restrict__ b2g, const float* __restrict__ b3g,
    const float* __restrict__ h1g, const float* __restrict__ v1g,
    float* __restrict__ s2pre) {
  extern __shared__ char smem[];
  float* b0 = (float*)smem; float* b1 = b0 + 64; float* b2 = b1 + 64;
  float* b3 = b2 + 64; float* mk = b3 + 256; float* y1 = mk + 64;   // [64][3]
  float* h1L = y1 + 64 * 3 + 4;                                     // [64][128]
  float* red = h1L + 64 * KD;                                       // 4*128
  h16* w0 = (h16*)(red + 512);
  h16* w1 = w0 + R1W0; h16* w2 = w1 + RW1; h16* w3 = w2 + RW2;
  h16* a0 = w0 + R2_BLOB;
  h16* hA = a0 + 64 * 32; h16* hB = hA + 64 * 64;
  h16* we2 = hB + 64 * 64;                                          // [64][256]

  int tid = threadIdx.x, row = blockIdx.x, eb = row * NN, b = row >> 6;

  if (tid == 0) tdm_copy_to_lds(wblob, w0, (unsigned)(R2_BLOB * 2));

  for (int idx = tid; idx < 64; idx += 256) {
    b0[idx] = b0g[idx]; b1[idx] = b1g[idx]; b2[idx] = b2g[idx];
    mk[idx] = maskv[eb + idx];
  }
  for (int idx = tid; idx < 256; idx += 256) b3[idx] = b3g[idx];
  for (int idx = tid; idx < 64 * 4; idx += 256) {
    int j = idx >> 2, q = idx & 3;
    v8h z = {};
    ((v8h*)a0)[idx] = (q == 0) ? *(const v8h*)(radf + (eb + j) * NBES) : z;
  }
  for (int idx = tid; idx < 64 * (KD / 4); idx += 256)
    ((float4*)h1L)[idx] = ((const float4*)(h1g + (size_t)(b * NN) * KD))[idx];
  for (int idx = tid; idx < 64 * 3; idx += 256) {
    int j = idx / 3, m = idx % 3;
    y1[idx] = yv[(eb + j) * 9 + 1 + m];
  }
  __builtin_amdgcn_s_wait_tensorcnt(0);
  __syncthreads();

  run_fc_layer(a0, 32, 32, w0, 64, b0, hA, 64, 1, nullptr, 0);
  run_fc_layer(hA, 64, 64, w1, 64, b1, hB, 64, 1, nullptr, 0);
  run_fc_layer(hB, 64, 64, w2, 64, b2, hA, 64, 1, nullptr, 0);
  run_fc_layer(hA, 64, 64, w3, 256, b3, we2, 256, 0, mk, 0);  // row-major

  // msg_a[k] = sum_j we2[j,k]*h1[b,j,k] ; msg_b[k] = sum_j we2[j,128+k]*inv[j,k]
  int k = tid & 127, hs = tid >> 7;
  float sa = 0.f, sb = 0.f;
  for (int j = hs * 32; j < hs * 32 + 32; ++j) {
    float invv = y1[j * 3 + 0] * v1g[((b * NN + j) * 3 + 0) * KD + k] +
                 y1[j * 3 + 1] * v1g[((b * NN + j) * 3 + 1) * KD + k] +
                 y1[j * 3 + 2] * v1g[((b * NN + j) * 3 + 2) * KD + k];
    sa += (float)we2[j * 256 + k] * h1L[j * KD + k];
    sb += (float)we2[j * 256 + 128 + k] * invv;
  }
  red[hs * KD + k] = sa;
  red[256 + hs * KD + k] = sb;
  __syncthreads();
  if (tid < KD)
    s2pre[row * KD + tid] =
        (red[tid] + red[KD + tid] + red[256 + tid] + red[256 + KD + tid]) * INV_AVG;
}

// ---------------------------------------------------------------------------
// Kernel 5: s2 = s2pre @ mix2 (WMMA, TDM-staged swizzled weights).
// ---------------------------------------------------------------------------
__global__ void k_mix2(const float* __restrict__ s2pre, const h16* __restrict__ mixsw,
                       float* __restrict__ s2g) {
  extern __shared__ char smem[];
  h16* Wm = (h16*)smem; h16* A = Wm + KD * KD;
  int tid = threadIdx.x, rb = blockIdx.x * 64;
  if (tid == 0) tdm_copy_to_lds(mixsw, Wm, (unsigned)(KD * KD * 2));
  for (int idx = tid; idx < 64 * KD; idx += 256) A[idx] = (h16)s2pre[rb * KD + idx];
  __builtin_amdgcn_s_wait_tensorcnt(0);
  __syncthreads();
  int l = lane_id(), n = l & 15, mb = (l < 16) ? 0 : 8;
  for (int t = wave_id(); t < 32; t += 8) {
    int rt = t >> 3, ct = t & 7;
    v8f acc = v8f_zero();
    acc = gemm_swz(acc, A + rt * 16 * KD, KD, Wm, 8, ct, KD);
#pragma unroll
    for (int v = 0; v < 8; ++v)
      s2g[(rb + rt * 16 + mb + v) * KD + ct * 16 + n] = acc[v];
  }
}

// ---------------------------------------------------------------------------
// Kernel 6: assemble padded f16 feature matrix X = [s1 | v1 | s2 | t | 0pad].
// ---------------------------------------------------------------------------
__global__ void k_feat(const float* __restrict__ s1g, const float* __restrict__ v1g,
                       const float* __restrict__ s2g, const float* __restrict__ timeg,
                       h16* __restrict__ X) {
  int gid = blockIdx.x * 256 + threadIdx.x;
  if (gid >= NNODE * FPAD) return;
  int row = gid / FPAD, c = gid % FPAD;
  float v;
  if (c < KD)            v = s1g[row * KD + c];
  else if (c < 4 * KD)   v = v1g[row * 3 * KD + (c - KD)];
  else if (c < 5 * KD)   v = s2g[row * KD + (c - 4 * KD)];
  else if (c == 5 * KD)  v = timeg[row >> 6];
  else                   v = 0.f;
  X[gid] = (h16)v;
}

// ---------------------------------------------------------------------------
// Kernel 7: streamed-K WMMA GEMM + bias + ReLU on pre-swizzled f16 weights.
// Block tile 64 x 256; K staged through LDS in 32-deep slabs (b128 copies).
// ---------------------------------------------------------------------------
__global__ void k_fc(const h16* __restrict__ Ain, int lda,
                     const h16* __restrict__ Wsw, const float* __restrict__ biasg,
                     int Ncols, int Kpad, h16* __restrict__ Out) {
  extern __shared__ char smem[];
  float* biasL = (float*)smem;
  h16* Wt = (h16*)(biasL + 256);   // 16 swizzled tiles = 8192 halves
  h16* At = Wt + 32 * 256;         // [64][32]
  int tid = threadIdx.x;
  int rb = blockIdx.x * 64, cb = blockIdx.y * 256;
  int nctg = Ncols >> 4, ctg0 = cb >> 4;
  for (int idx = tid; idx < 256; idx += 256) biasL[idx] = biasg[cb + idx];
  int l = lane_id(), n = l & 15, mb = (l < 16) ? 0 : 8;
  int w = wave_id(), rt = w & 3, cg = (w >> 2) * 8;
  v8f acc[8];
#pragma unroll
  for (int ci = 0; ci < 8; ++ci) acc[ci] = v8f_zero();
  for (int kc = 0; kc < (Kpad >> 5); ++kc) {
    __syncthreads();
    const v8h* srcW = (const v8h*)(Wsw + (size_t)(kc * nctg + ctg0) * 512);
    for (int idx = tid; idx < 1024; idx += 256) ((v8h*)Wt)[idx] = srcW[idx];
    for (int idx = tid; idx < 256; idx += 256) {
      int r = idx >> 2, q = idx & 3;
      ((v8h*)At)[idx] = *(const v8h*)(Ain + (size_t)(rb + r) * lda + kc * 32 + q * 8);
    }
    if (kc + 1 < (Kpad >> 5))
      __builtin_prefetch(Wsw + (size_t)((kc + 1) * nctg + ctg0) * 512, 0, 1);
    __syncthreads();
    v16h a = frag_a_vec(At + rt * 16 * 32, 32);
#pragma unroll
    for (int ci = 0; ci < 8; ++ci)
      acc[ci] = wmma_f16(a, frag_b_swz(Wt + (cg + ci) * 512), acc[ci]);
  }
#pragma unroll
  for (int ci = 0; ci < 8; ++ci)
#pragma unroll
    for (int v = 0; v < 8; ++v) {
      int row = rb + rt * 16 + mb + v, col = cb + (cg + ci) * 16 + n;
      float x = acc[ci][v] + biasL[(cg + ci) * 16 + n];
      Out[row * Ncols + col] = (h16)fmaxf(x, 0.f);
    }
}

// ---------------------------------------------------------------------------
// Kernel 8: final 512 -> 3 projection (N too small for WMMA; VALU dot).
// ---------------------------------------------------------------------------
__global__ void k_fc2(const h16* __restrict__ H2, const float* __restrict__ w2g,
                      const float* __restrict__ b2g, float* __restrict__ out) {
  __shared__ float wL[HML * 3];
  int tid = threadIdx.x;
  for (int idx = tid; idx < HML * 3; idx += 256) wL[idx] = w2g[idx];
  __syncthreads();
  int gid = blockIdx.x * 256 + tid;
  if (gid >= NNODE * 3) return;
  int row = gid / 3, c = gid % 3;
  float s = b2g[c];
  for (int k = 0; k < HML; ++k) s += (float)H2[row * HML + k] * wL[k * 3 + c];
  out[gid] = s;
}

// ---------------------------------------------------------------------------
// Host launcher.
// ---------------------------------------------------------------------------
static constexpr size_t SM_P1 =
    640 * 4 + (size_t)(R1_BLOB + 64 * 32 + 64 * 64 * 2 + 64 * 384 + 3 * 16 * 64) * 2;
static constexpr size_t SM_P2 =
    (640 + 192 + 4 + 64 * KD + 512) * 4 +
    (size_t)(R2_BLOB + 64 * 32 + 64 * 64 * 2 + 64 * 256) * 2;
static constexpr size_t SM_MIX1 = (size_t)(3 * KD * KD + 64 * KD * 2 + 192 * KD) * 2;
static constexpr size_t SM_MIX2 = (size_t)(KD * KD + 64 * KD) * 2;
static constexpr size_t SM_FC   = 256 * 4 + (size_t)(32 * 256 + 64 * 32) * 2;

extern "C" void kernel_launch(void* const* d_in, const int* in_sizes, int n_in,
                              void* d_out, int out_size, void* d_ws, size_t ws_size,
                              hipStream_t stream) {
  (void)in_sizes; (void)n_in; (void)out_size; (void)ws_size;
  const float* pos   = (const float*)d_in[0];
  const float* timeg = (const float*)d_in[1];
  const float* cell  = (const float*)d_in[2];
  const float* wemb  = (const float*)d_in[3];
  const float* r1w0 = (const float*)d_in[4];  const float* r1b0 = (const float*)d_in[5];
  const float* r1w1 = (const float*)d_in[6];  const float* r1b1 = (const float*)d_in[7];
  const float* r1w2 = (const float*)d_in[8];  const float* r1b2 = (const float*)d_in[9];
  const float* r1w3 = (const float*)d_in[10]; const float* r1b3 = (const float*)d_in[11];
  const float* m1l0 = (const float*)d_in[12];
  const float* m1l1 = (const float*)d_in[13];
  const float* m1l2 = (const float*)d_in[14];
  const float* r2w0 = (const float*)d_in[15]; const float* r2b0 = (const float*)d_in[16];
  const float* r2w1 = (const float*)d_in[17]; const float* r2b1 = (const float*)d_in[18];
  const float* r2w2 = (const float*)d_in[19]; const float* r2b2 = (const float*)d_in[20];
  const float* r2w3 = (const float*)d_in[21]; const float* r2b3 = (const float*)d_in[22];
  const float* mix2 = (const float*)d_in[23];
  const float* mw0 = (const float*)d_in[24]; const float* mb0 = (const float*)d_in[25];
  const float* mw1 = (const float*)d_in[26]; const float* mb1 = (const float*)d_in[27];
  const float* mw2 = (const float*)d_in[28]; const float* mb2 = (const float*)d_in[29];

  char* ws = (char*)d_ws;
  size_t off = 0;
  auto carve = [&](size_t bytes) {
    char* p = ws + off;
    off = (off + bytes + 255) & ~size_t(255);
    return p;
  };
  h16*   radf  = (h16*)carve((size_t)M_EDGE * NBES * 2);
  float* yv    = (float*)carve((size_t)M_EDGE * 9 * 4);
  float* maskv = (float*)carve((size_t)M_EDGE * 4);
  float* msg   = (float*)carve((size_t)NNODE * 9 * KD * 4);
  float* s1    = (float*)carve((size_t)NNODE * KD * 4);
  float* h1    = (float*)carve((size_t)NNODE * KD * 4);
  float* v1    = (float*)carve((size_t)NNODE * 3 * KD * 4);
  float* s2pre = (float*)carve((size_t)NNODE * KD * 4);
  float* s2    = (float*)carve((size_t)NNODE * KD * 4);
  h16*   X     = (h16*)carve((size_t)NNODE * FPAD * 2);
  h16*   H1    = (h16*)carve((size_t)NNODE * HML * 2);
  h16*   H2    = (h16*)carve((size_t)NNODE * HML * 2);
  h16*   r1b   = (h16*)carve((size_t)R1_BLOB * 2);
  h16*   r2b   = (h16*)carve((size_t)R2_BLOB * 2);
  h16*   mixb  = (h16*)carve((size_t)3 * KD * KD * 2);
  h16*   mix2b = (h16*)carve((size_t)KD * KD * 2);
  h16*   fc0b  = (h16*)carve((size_t)FPAD * HML * 2);
  h16*   fc1b  = (h16*)carve((size_t)HML * HML * 2);

  auto swz = [&](h16* dst, const float* src, int Ksz, int Krows, int Ncols) {
    int total = Ksz * Ncols;
    k_swz<<<(total + 255) / 256, 256, 0, stream>>>(dst, src, Ksz, Krows, Ncols);
  };
  // pre-convert + swizzle all weights to f16 fragment order (once)
  swz(r1b, r1w0, 32, NBES, 64);
  swz(r1b + R1W0, r1w1, 64, 64, 64);
  swz(r1b + R1W0 + RW1, r1w2, 64, 64, 64);
  swz(r1b + R1W0 + RW1 + RW2, r1w3, 64, 64, 384);
  swz(r2b, r2w0, 32, NBES, 64);
  swz(r2b + R1W0, r2w1, 64, 64, 64);
  swz(r2b + R1W0 + RW1, r2w2, 64, 64, 64);
  swz(r2b + R1W0 + RW1 + RW2, r2w3, 64, 64, 256);
  swz(mixb, m1l0, KD, KD, KD);
  swz(mixb + KD * KD, m1l1, KD, KD, KD);
  swz(mixb + 2 * KD * KD, m1l2, KD, KD, KD);
  swz(mix2b, mix2, KD, KD, KD);
  swz(fc0b, mw0, FPAD, 641, HML);
  swz(fc1b, mw1, HML, HML, HML);

  k_geom<<<M_EDGE / 256, 256, 0, stream>>>(pos, cell, radf, yv, maskv);
  k_edge_pass1<<<NNODE, 256, SM_P1, stream>>>(radf, yv, maskv, r1b,
      r1b0, r1b1, r1b2, r1b3, wemb, msg);
  k_mix1<<<NNODE / 64, 256, SM_MIX1, stream>>>(msg, mixb, s1, h1, v1);
  k_pass2<<<NNODE, 256, SM_P2, stream>>>(radf, yv, maskv, r2b,
      r2b0, r2b1, r2b2, r2b3, h1, v1, s2pre);
  k_mix2<<<NNODE / 64, 256, SM_MIX2, stream>>>(s2pre, mix2b, s2);
  k_feat<<<(NNODE * FPAD) / 256, 256, 0, stream>>>(s1, v1, s2, timeg, X);
  k_fc<<<dim3(NNODE / 64, HML / 256), 256, SM_FC, stream>>>(X, FPAD, fc0b, mb0,
                                                            HML, FPAD, H1);
  k_fc<<<dim3(NNODE / 64, HML / 256), 256, SM_FC, stream>>>(H1, HML, fc1b, mb1,
                                                            HML, HML, H2);
  k_fc2<<<(NNODE * 3 + 255) / 256, 256, 0, stream>>>(H2, mw2, mb2, (float*)d_out);
}
